// Model_50929722196689
// MI455X (gfx1250) — matrix operand order repaired and checked by reading, not compiled, not run on hardware
//
#include <hip/hip_runtime.h>
#include <math.h>

// ---------------- problem constants ----------------
#define BB   64
#define SS   361
#define CC   512
#define HH   8
#define DD   64
#define FFND 1536
#define TT   16
#define MROWS (BB*SS)   // 23104, divisible by 64
#define QTILES 23       // ceil(361/16)

// ---------------- CDNA5 WMMA types ----------------
typedef __attribute__((ext_vector_type(16))) __bf16 v16bf;
typedef __attribute__((ext_vector_type(8)))  float  v8f;

union Frag {
  v16bf b;
  uint4 q[2];
  unsigned short e[16];
};

#define WMMA_BF16(a_, b_, c_) \
  __builtin_amdgcn_wmma_f32_16x16x32_bf16(false, (a_), false, (b_), (short)0, (c_), false, false)

__device__ __forceinline__ unsigned short f2bf(float f) {
  unsigned u = __float_as_uint(f);
  u += 0x7fffu + ((u >> 16) & 1u);          // round-to-nearest-even
  return (unsigned short)(u >> 16);
}
__device__ __forceinline__ float bf2f(unsigned short h) {
  return __uint_as_float(((unsigned)h) << 16);
}
__device__ __forceinline__ float gelu_exact(float x) {
  return 0.5f * x * (1.0f + erff(x * 0.70710678118654752f));
}

// ---------------- weight transpose + bf16 convert: wt[n][k] = bf16(w[k][n]) ----------------
__global__ void wtrans_kernel(const float* __restrict__ w, unsigned short* __restrict__ wt,
                              int K, int N) {
  size_t i = (size_t)blockIdx.x * 256 + threadIdx.x;
  if (i >= (size_t)K * N) return;
  int k = (int)(i % K), n = (int)(i / K);
  wt[i] = f2bf(w[(size_t)k * N + n]);
}

// ---------------- templates[s,q,:] = gelu(feats @ w1 + b1) @ w2 + b2 ----------------
__global__ __launch_bounds__(128) void templates_kernel(
    const float* __restrict__ freqs, const float* __restrict__ w1, const float* __restrict__ b1,
    const float* __restrict__ w2, const float* __restrict__ b2, float* __restrict__ tmpl) {
  int idx = blockIdx.x * blockDim.x + threadIdx.x;
  if (idx >= SS * SS) return;
  int i = idx / SS, j = idx % SS;
  float dr = (float)(i / 19 - j / 19);
  float dc = (float)(i % 19 - j % 19);
  float combo[4] = {dr, dc, dr + dc, dr - dc};
  float h[64];
  for (int u = 0; u < 64; ++u) h[u] = b1[u];
  for (int f = 0; f < 8; ++f) {
    float fr = freqs[f];
    for (int cp = 0; cp < 4; ++cp) {
      float ang = combo[cp] * fr;
      float sv = sinf(ang), cv = cosf(ang);
      int fi = f * 8 + cp * 2;
      for (int u = 0; u < 64; ++u) h[u] += sv * w1[fi * 64 + u];
      for (int u = 0; u < 64; ++u) h[u] += cv * w1[(fi + 1) * 64 + u];
    }
  }
  float o[16];
  for (int t = 0; t < 16; ++t) o[t] = b2[t];
  for (int u = 0; u < 64; ++u) {
    float hv = gelu_exact(h[u]);
    for (int t = 0; t < 16; ++t) o[t] += hv * w2[u * 16 + t];
  }
  for (int t = 0; t < 16; ++t) tmpl[(size_t)idx * 16 + t] = o[t];
}

// ---------------- RMSNorm: one block per row, output bf16 ----------------
__global__ __launch_bounds__(256) void rmsnorm_kernel(const float* __restrict__ x,
                                                      const float* __restrict__ wgt,
                                                      unsigned short* __restrict__ out) {
  __shared__ float red[256];
  int row = blockIdx.x, t = threadIdx.x;
  const float* xr = x + (size_t)row * CC;
  float v0 = xr[t], v1 = xr[t + 256];
  red[t] = v0 * v0 + v1 * v1;
  __syncthreads();
  for (int off = 128; off > 0; off >>= 1) {
    if (t < off) red[t] += red[t + off];
    __syncthreads();
  }
  float inv = rsqrtf(red[0] / (float)CC + 1e-6f);
  out[(size_t)row * CC + t]       = f2bf(v0 * inv * wgt[t]);
  out[(size_t)row * CC + t + 256] = f2bf(v1 * inv * wgt[t + 256]);
}

// ---------------- xsum[b,c] = sum_s xn[b,s,c] ----------------
__global__ void xsum_kernel(const unsigned short* __restrict__ xn, float* __restrict__ xsum) {
  int i = blockIdx.x * 256 + threadIdx.x;   // b*512 + c
  int b = i >> 9, c = i & 511;
  const unsigned short* p = xn + (size_t)b * SS * CC + c;
  float a = 0.f;
  for (int s = 0; s < SS; ++s) a += bf2f(p[(size_t)s * CC]);
  xsum[i] = a;
}

// ---------------- pooled -> gelu/LN chain -> z[b, H*T] ----------------
__global__ __launch_bounds__(128) void zchain_kernel(
    const float* __restrict__ xsum, const float* __restrict__ gproj1,
    const float* __restrict__ gproj2, const float* __restrict__ gn1g, const float* __restrict__ gn1b,
    const float* __restrict__ gproj3, const float* __restrict__ gn2g, const float* __restrict__ gn2b,
    float* __restrict__ z) {
  __shared__ float pooled[16], z2[16], z3[128], st[2];
  int b = blockIdx.x, t = threadIdx.x;
  if (t < 16) {
    float a = 0.f;
    const float* xs = xsum + (size_t)b * CC;
    for (int c = 0; c < CC; ++c) a += xs[c] * gproj1[c * 16 + t];
    pooled[t] = a / (float)SS;
  }
  __syncthreads();
  if (t < 16) {
    float a = 0.f;
    for (int i = 0; i < 16; ++i) a += pooled[i] * gproj2[i * 16 + t];
    z2[t] = gelu_exact(a);
  }
  __syncthreads();
  if (t == 0) {
    float mu = 0.f; for (int i = 0; i < 16; ++i) mu += z2[i]; mu *= (1.f/16.f);
    float var = 0.f; for (int i = 0; i < 16; ++i) { float d = z2[i]-mu; var += d*d; } var *= (1.f/16.f);
    st[0] = mu; st[1] = rsqrtf(var + 1e-5f);
  }
  __syncthreads();
  if (t < 16) z2[t] = (z2[t] - st[0]) * st[1] * gn1g[t] + gn1b[t];
  __syncthreads();
  {
    float a = 0.f;
    for (int i = 0; i < 16; ++i) a += z2[i] * gproj3[i * 128 + t];
    z3[t] = gelu_exact(a);
  }
  __syncthreads();
  if (t == 0) {
    float mu = 0.f; for (int i = 0; i < 128; ++i) mu += z3[i]; mu *= (1.f/128.f);
    float var = 0.f; for (int i = 0; i < 128; ++i) { float d = z3[i]-mu; var += d*d; } var *= (1.f/128.f);
    st[0] = mu; st[1] = rsqrtf(var + 1e-5f);
  }
  __syncthreads();
  z[(size_t)b * 128 + t] = (z3[t] - st[0]) * st[1] * gn2g[t] + gn2b[t];
}

// ---------------- bias[bh,q,k] = z[bh,:].templates[q,k,:]  (templates in regs, bh innermost) ----------------
__global__ __launch_bounds__(384) void bias_kernel(const float* __restrict__ z,
                                                   const float* __restrict__ tmpl,
                                                   float* __restrict__ bias) {
  __shared__ float zsh[BB * HH * TT];   // 8192 floats = 32 KB
  int q = blockIdx.x;
  for (int i = threadIdx.x; i < BB * HH * TT; i += 384) zsh[i] = z[i];
  __syncthreads();
  int k = threadIdx.x;
  if (k >= SS) return;
  float tv[16];
  const float* tp = tmpl + ((size_t)q * SS + k) * TT;
  for (int t = 0; t < 16; ++t) tv[t] = tp[t];
  for (int bh = 0; bh < BB * HH; ++bh) {
    const float* zz = &zsh[bh * 16];
    float a = 0.f;
    for (int t = 0; t < 16; ++t) a += zz[t] * tv[t];
    bias[((size_t)bh * SS + q) * SS + k] = a;
  }
}

// ---------------- generic bf16 WMMA GEMM: out = A[M,K] @ BT[N,K]^T, 64x64 wave tiles ----------------
// mode 0: outF = acc ; mode 1: outF = acc + aux ; mode 2: outB = bf16(acc * silu(aux))
__global__ __launch_bounds__(256) void gemm_bf16_wmma(
    const unsigned short* __restrict__ A, const unsigned short* __restrict__ BT,
    float* __restrict__ outF, unsigned short* __restrict__ outB,
    const float* __restrict__ aux, int N, int K, int mode) {
  __shared__ unsigned short aLds[64 * 32];
  const int t = threadIdx.x;
  const int w = t >> 5, l = t & 31, hl = l >> 4, ln = l & 15;
  const int mbase = blockIdx.x * 64;
  const int nbase = blockIdx.y * 512 + w * 64;

  const v8f vz = {0.f,0.f,0.f,0.f,0.f,0.f,0.f,0.f};
  v8f acc[4][4];
  for (int f = 0; f < 4; ++f) for (int s = 0; s < 4; ++s) acc[f][s] = vz;

  for (int k0 = 0; k0 < K; k0 += 32) {
    __syncthreads();
    {
      int row = t >> 2, cb = t & 3;
      const unsigned short* src = A + (size_t)(mbase + row) * K + k0 + cb * 8;
      reinterpret_cast<uint4*>(aLds)[t] = *reinterpret_cast<const uint4*>(src);
      if (k0 + 32 < K) __builtin_prefetch(src + 32, 0, 1);   // global_prefetch
    }
    __syncthreads();
    Frag a[4];
    for (int f = 0; f < 4; ++f) {
      const uint4* lp = reinterpret_cast<const uint4*>(&aLds[(f * 16 + ln) * 32]);
      a[f].q[0] = lp[2 * hl];      // K = hl*16 + 0..7
      a[f].q[1] = lp[2 * hl + 1];  // K = hl*16 + 8..15
    }
    for (int s = 0; s < 4; ++s) {
      Frag bfr;
      const int n = nbase + s * 16 + ln;
      const uint4* bp = reinterpret_cast<const uint4*>(BT + (size_t)n * K + k0 + hl * 16);
      bfr.q[0] = bp[0]; bfr.q[1] = bp[1];     // K = hl*16 + 0..15
      for (int f = 0; f < 4; ++f) acc[f][s] = WMMA_BF16(a[f].b, bfr.b, acc[f][s]);
    }
  }
  for (int f = 0; f < 4; ++f)
    for (int s = 0; s < 4; ++s)
      for (int r = 0; r < 8; ++r) {
        int row = mbase + f * 16 + hl * 8 + r;
        int col = nbase + s * 16 + ln;
        size_t idx = (size_t)row * N + col;
        float v = acc[f][s][r];
        if (mode == 1) v += aux[idx];
        if (mode == 2) {
          float g = aux[idx];
          v *= g * (1.0f / (1.0f + __expf(-g)));   // silu(gemm1) * gate
          outB[idx] = f2bf(v);
        } else {
          outF[idx] = v;
        }
      }
}

// ---------------- RoPE + pack [B,S,H,D] fp32 -> [B,H,S,D] bf16 ----------------
__global__ void rope_pack_kernel(const float* __restrict__ qf, const float* __restrict__ kf,
                                 const float* __restrict__ vf, const float* __restrict__ cosb,
                                 const float* __restrict__ sinb, unsigned short* __restrict__ qb,
                                 unsigned short* __restrict__ kb, unsigned short* __restrict__ vb) {
  size_t i = (size_t)blockIdx.x * 256 + threadIdx.x;   // grid sized exactly
  int d = (int)(i & 63);
  size_t r = i >> 6;
  int h = (int)(r & 7);
  size_t r2 = r >> 3;
  int s = (int)(r2 % SS);
  int b = (int)(r2 / SS);
  float c = cosb[s * DD + d], sn = sinb[s * DD + d];
  size_t pair = (d < 32) ? i + 32 : i - 32;
  float sgn = (d < 32) ? -1.f : 1.f;
  float qv = qf[i] * c + sgn * qf[pair] * sn;
  float kv = kf[i] * c + sgn * kf[pair] * sn;
  size_t oi = (((size_t)b * HH + h) * SS + s) * DD + d;
  qb[oi] = f2bf(qv);
  kb[oi] = f2bf(kv);
  vb[oi] = f2bf(vf[i]);
}

// ---------------- flash attention: one wave per (b,h,qtile), WMMA QK^T and PV ----------------
__global__ __launch_bounds__(256) void attn_kernel(
    const unsigned short* __restrict__ qb, const unsigned short* __restrict__ kb,
    const unsigned short* __restrict__ vb, const float* __restrict__ bias,
    unsigned short* __restrict__ attn_out) {
  __shared__ unsigned short plds[8][256];   // 16x16 P tile per wave
  const int t = threadIdx.x, w = t >> 5, l = t & 31, hl = l >> 4, ln = l & 15;
  const int gw = blockIdx.x * 8 + w;        // wave id, grid sized exactly
  const int qt = gw % QTILES;
  const int bh = gw / QTILES;               // b*H + h
  const int q0 = qt * 16;
  const size_t base = (size_t)bh * SS * DD;
  const v8f vz = {0.f,0.f,0.f,0.f,0.f,0.f,0.f,0.f};
  const uint4 z4 = make_uint4(0u, 0u, 0u, 0u);

  // Q A-fragments for d-chunks 0..31 and 32..63
  Frag aQ[2];
  {
    int qrow = q0 + ln;
    if (qrow < SS) {
      const uint4* qp = reinterpret_cast<const uint4*>(qb + base + (size_t)qrow * DD);
      aQ[0].q[0] = qp[2 * hl];     aQ[0].q[1] = qp[2 * hl + 1];
      aQ[1].q[0] = qp[4 + 2 * hl]; aQ[1].q[1] = qp[5 + 2 * hl];
    } else {
      aQ[0].q[0] = z4; aQ[0].q[1] = z4; aQ[1].q[0] = z4; aQ[1].q[1] = z4;
    }
  }
  float mrow[8], lrow[8];
  for (int r = 0; r < 8; ++r) { mrow[r] = -3.0e38f; lrow[r] = 0.f; }
  v8f o[4] = {vz, vz, vz, vz};
  const float* biasp = bias + (size_t)bh * SS * SS;

  for (int kt = 0; kt < QTILES; ++kt) {
    const int k0 = kt * 16;
    const int key = k0 + ln;
    Frag bK[2];
    if (key < SS) {
      const uint4* kp = reinterpret_cast<const uint4*>(kb + base + (size_t)key * DD);
      bK[0].q[0] = kp[hl * 2];     bK[0].q[1] = kp[hl * 2 + 1];
      bK[1].q[0] = kp[4 + hl * 2]; bK[1].q[1] = kp[4 + hl * 2 + 1];
    } else {
      bK[0].q[0] = z4; bK[0].q[1] = z4; bK[1].q[0] = z4; bK[1].q[1] = z4;
    }
    v8f s = vz;
    s = WMMA_BF16(aQ[0].b, bK[0].b, s);
    s = WMMA_BF16(aQ[1].b, bK[1].b, s);

    float p[8];
    for (int r = 0; r < 8; ++r) {
      int qr = q0 + hl * 8 + r;
      float val;
      if (key < SS && qr < SS) val = s[r] * 0.125f + biasp[(size_t)qr * SS + key];
      else val = -1.0e30f;
      float mx = val;
      for (int off = 1; off < 16; off <<= 1) mx = fmaxf(mx, __shfl_xor(mx, off, 32));
      float mnew = fmaxf(mrow[r], mx);
      float pe = __expf(val - mnew);
      float sum = pe;
      for (int off = 1; off < 16; off <<= 1) sum += __shfl_xor(sum, off, 32);
      float fac = __expf(mrow[r] - mnew);
      lrow[r] = lrow[r] * fac + sum;
      mrow[r] = mnew;
      for (int dt = 0; dt < 4; ++dt) o[dt][r] *= fac;
      p[r] = pe;
    }
    __syncthreads();   // uniform: guards previous iteration's LDS reads
    for (int r = 0; r < 8; ++r) plds[w][(hl * 8 + r) * 16 + ln] = f2bf(p[r]);
    __syncthreads();
    // P as A-fragment: only K=0..15 valid, upper half zero-padded
    Frag aP;
    aP.q[0] = hl ? z4 : reinterpret_cast<const uint4*>(&plds[w][ln * 16])[0];
    aP.q[1] = hl ? z4 : reinterpret_cast<const uint4*>(&plds[w][ln * 16])[1];
    for (int dt = 0; dt < 4; ++dt) {
      Frag bV;
      if (hl == 0) {
        for (int e = 0; e < 16; ++e) {
          int kk = k0 + e;
          bV.e[e] = (kk < SS) ? vb[base + (size_t)kk * DD + dt * 16 + ln] : (unsigned short)0;
        }
      } else {
        bV.q[0] = z4; bV.q[1] = z4;   // zero-padded keys 16..31
      }
      o[dt] = WMMA_BF16(aP.b, bV.b, o[dt]);
    }
  }
  const int b = bh >> 3, h = bh & 7;
  for (int dt = 0; dt < 4; ++dt)
    for (int r = 0; r < 8; ++r) {
      int qr = q0 + hl * 8 + r;
      if (qr < SS) {
        float ov = o[dt][r] / lrow[r];
        attn_out[((size_t)b * SS + qr) * CC + h * DD + dt * 16 + ln] = f2bf(ov);
      }
    }
}

// =====================================================================
extern "C" void kernel_launch(void* const* d_in, const int* in_sizes, int n_in,
                              void* d_out, int out_size, void* d_ws, size_t ws_size,
                              hipStream_t stream) {
  const float* x        = (const float*)d_in[0];
  const float* rope_cos = (const float*)d_in[1];
  const float* rope_sin = (const float*)d_in[2];
  const float* gab_freq = (const float*)d_in[3];
  const float* gmlp_w1  = (const float*)d_in[4];
  const float* gmlp_b1  = (const float*)d_in[5];
  const float* gmlp_w2  = (const float*)d_in[6];
  const float* gmlp_b2  = (const float*)d_in[7];
  const float* norm1_w  = (const float*)d_in[8];
  const float* wq       = (const float*)d_in[9];
  const float* wk       = (const float*)d_in[10];
  const float* wv       = (const float*)d_in[11];
  const float* wo       = (const float*)d_in[12];
  const float* gproj1   = (const float*)d_in[13];
  const float* gproj2   = (const float*)d_in[14];
  const float* gn1_g    = (const float*)d_in[15];
  const float* gn1_b    = (const float*)d_in[16];
  const float* gproj3   = (const float*)d_in[17];
  const float* gn2_g    = (const float*)d_in[18];
  const float* gn2_b    = (const float*)d_in[19];
  const float* norm2_w  = (const float*)d_in[20];
  const float* ffn_w1   = (const float*)d_in[21];
  const float* ffn_wg   = (const float*)d_in[22];
  const float* ffn_w2   = (const float*)d_in[23];

  char* ws = (char*)d_ws;
  const size_t MB = 1ull << 20;
  // --- workspace layout (lifetime-overlapped; peak ~453 MB) ---
  unsigned short* wqT = (unsigned short*)(ws + 0);
  unsigned short* wkT = (unsigned short*)(ws + 512 * 1024);
  unsigned short* wvT = (unsigned short*)(ws + 1 * MB);
  unsigned short* woT = (unsigned short*)(ws + 1 * MB + 512 * 1024);
  unsigned short* w1T = (unsigned short*)(ws + 2 * MB);
  unsigned short* wgT = (unsigned short*)(ws + 2 * MB + 1536 * 1024);
  unsigned short* w2T = (unsigned short*)(ws + 5 * MB);
  float* tmpl  = (float*)(ws + 8 * MB);          // 8.34 MB
  float* zbuf  = (float*)(ws + 17 * MB);         // 32 KB
  float* xsum  = (float*)(ws + 17 * MB + 64 * 1024);
  unsigned short* xn   = (unsigned short*)(ws + 18 * MB);   // 23.7 MB (reused as xn2)
  float* x1    = (float*)(ws + 42 * MB);         // 47.3 MB
  unsigned short* atto = (unsigned short*)(ws + 90 * MB);   // 23.7 MB
  unsigned short* qbuf = (unsigned short*)(ws + 114 * MB);  // qb/kb/vb each 23.7 MB
  unsigned short* kbuf = (unsigned short*)(ws + 138 * MB);
  unsigned short* vbuf = (unsigned short*)(ws + 162 * MB);
  unsigned short* h1   = (unsigned short*)(ws + 114 * MB);  // reuses qb/kb/vb after attention
  // BIG region (186 MB): qf/kf/vf  -> then bias -> then ffa
  float* qf   = (float*)(ws + 186 * MB);
  float* kf   = (float*)(ws + 234 * MB);
  float* vf   = (float*)(ws + 282 * MB);
  float* bias = (float*)(ws + 186 * MB);         // 267 MB (after rope_pack frees qf/kf/vf)
  float* ffa  = (float*)(ws + 186 * MB);         // 142 MB (after attention frees bias)
  (void)ws_size; (void)in_sizes; (void)n_in; (void)out_size;

  // 1) weight converts/transposes (bf16, [N][K])
  wtrans_kernel<<<(512 * 512 + 255) / 256, 256, 0, stream>>>(wq, wqT, 512, 512);
  wtrans_kernel<<<(512 * 512 + 255) / 256, 256, 0, stream>>>(wk, wkT, 512, 512);
  wtrans_kernel<<<(512 * 512 + 255) / 256, 256, 0, stream>>>(wv, wvT, 512, 512);
  wtrans_kernel<<<(512 * 512 + 255) / 256, 256, 0, stream>>>(wo, woT, 512, 512);
  wtrans_kernel<<<(512 * 1536 + 255) / 256, 256, 0, stream>>>(ffn_w1, w1T, 512, 1536);
  wtrans_kernel<<<(512 * 1536 + 255) / 256, 256, 0, stream>>>(ffn_wg, wgT, 512, 1536);
  wtrans_kernel<<<(1536 * 512 + 255) / 256, 256, 0, stream>>>(ffn_w2, w2T, 1536, 512);

  // 2) templates
  templates_kernel<<<(SS * SS + 127) / 128, 128, 0, stream>>>(gab_freq, gmlp_w1, gmlp_b1,
                                                              gmlp_w2, gmlp_b2, tmpl);
  // 3) RMSNorm1 -> xn (bf16), column sums, z-chain
  rmsnorm_kernel<<<MROWS, 256, 0, stream>>>(x, norm1_w, xn);
  xsum_kernel<<<(BB * CC) / 256, 256, 0, stream>>>(xn, xsum);
  zchain_kernel<<<BB, 128, 0, stream>>>(xsum, gproj1, gproj2, gn1_g, gn1_b,
                                        gproj3, gn2_g, gn2_b, zbuf);
  // 4) QKV projections (WMMA)
  dim3 g512(MROWS / 64, 1), g1536(MROWS / 64, 3);
  gemm_bf16_wmma<<<g512, 256, 0, stream>>>(xn, wqT, qf, nullptr, nullptr, 512, 512, 0);
  gemm_bf16_wmma<<<g512, 256, 0, stream>>>(xn, wkT, kf, nullptr, nullptr, 512, 512, 0);
  gemm_bf16_wmma<<<g512, 256, 0, stream>>>(xn, wvT, vf, nullptr, nullptr, 512, 512, 0);
  // 5) RoPE + pack to [B,H,S,D] bf16 (frees qf/kf/vf region)
  rope_pack_kernel<<<(BB * SS * CC) / 256, 256, 0, stream>>>(qf, kf, vf, rope_cos, rope_sin,
                                                             qbuf, kbuf, vbuf);
  // 6) bias materialization (templates read once from L2)
  bias_kernel<<<SS, 384, 0, stream>>>(zbuf, tmpl, bias);
  // 7) flash attention (WMMA QK^T and PV)
  attn_kernel<<<(BB * HH * QTILES) / 8, 256, 0, stream>>>(qbuf, kbuf, vbuf, bias, atto);
  // 8) output projection + residual -> x1
  gemm_bf16_wmma<<<g512, 256, 0, stream>>>(atto, woT, x1, nullptr, x, 512, 512, 1);
  // 9) RMSNorm2 -> xn2 (reuses xn region)
  rmsnorm_kernel<<<MROWS, 256, 0, stream>>>(x1, norm2_w, xn);
  // 10) FFN: ffa = xn2@w1 ; h1 = silu(ffa)*(xn2@wgate) (bf16, fused epilogue)
  gemm_bf16_wmma<<<g1536, 256, 0, stream>>>(xn, w1T, ffa, nullptr, nullptr, 1536, 512, 0);
  gemm_bf16_wmma<<<g1536, 256, 0, stream>>>(xn, wgT, nullptr, h1, ffa, 1536, 512, 2);
  // 11) down-projection + residual -> d_out
  gemm_bf16_wmma<<<g512, 256, 0, stream>>>(h1, w2T, (float*)d_out, nullptr, x1, 512, 1536, 1);
}
